// BayesianKalmanNet_85392539779586
// MI455X (gfx1250) — matrix operand order repaired and checked by reading, not compiled, not run on hardware
//
#include <hip/hip_runtime.h>
#include <hip/hip_bf16.h>
#include <math.h>

// ---------------- problem constants ----------------
#define S_DIM 8
#define O_DIM 4
#define HID   800
#define BATCH 128
#define SEQ   32
#define NS    16
#define MROWS (NS * BATCH)        // 2048
#define N3    (3 * HID)           // 2400
#define KEEP_INV 1.25f            // 1/(1-0.2)

// ---------------- WMMA types ----------------
typedef __attribute__((ext_vector_type(16))) __bf16 v16bf;
typedef __attribute__((ext_vector_type(8)))  __bf16 v8bf;
typedef __attribute__((ext_vector_type(8)))  float  v8f;

union Frag { v16bf v; v8bf h[2]; };
union Acc  { v8f  v; float f[8]; };

// deterministic dropout mask: returns 0 or 1/keep
__device__ __forceinline__ float dropmask(unsigned a, unsigned b, unsigned c) {
    unsigned x = a * 0x9E3779B1u ^ (b + 0x7F4A7C15u) * 0x85EBCA77u
               ^ (c + 0x165667B1u) * 0xC2B2AE3Du;
    x ^= x >> 16; x *= 0x7FEB352Du;
    x ^= x >> 15; x *= 0x846CA68Bu;
    x ^= x >> 16;
    return (x <= 0xCCCCCCCCu) ? KEEP_INV : 0.0f;   // keep prob 0.8
}

// A fragment: 16x32 (MxK) bf16. baseU is wave-uniform (tile base), laneOff = (lane&15)*ld + (lane>>4)*8.
__device__ __forceinline__ v16bf loadFragA(const __bf16* baseU, int laneOff, int k0) {
    const __bf16* p = baseU + laneOff + k0;
    Frag f;
    f.h[0] = *(const v8bf*)(p);
    f.h[1] = *(const v8bf*)(p + 16);
    return f.v;
}

// B fragment: 32x16 (KxN) where B[k][n] = W[n][k], W row-major stride ld.
__device__ __forceinline__ v16bf loadFragB(const __bf16* baseU, int laneOff, int k0) {
    const __bf16* p = baseU + laneOff + k0;
    Frag f;
    f.h[0] = *(const v8bf*)(p);
    f.h[1] = *(const v8bf*)(p + 16);
    return f.v;
}

// full fragment set for one GRU k-step
struct FragSet {
    v16bf fa, fh, b0, b1, b2, b3, b4, b5;
};

__device__ __forceinline__ FragSet load_all(const __bf16* aBase, const __bf16* hBase,
                                            const __bf16* irBase, const __bf16* izBase,
                                            const __bf16* inBase, const __bf16* hrBase,
                                            const __bf16* hzBase, const __bf16* hnBase,
                                            int offA, int offB, int k0) {
    FragSet f;
    f.fa = loadFragA(aBase, offA, k0);
    f.fh = loadFragA(hBase, offA, k0);
    f.b0 = loadFragB(irBase, offB, k0);
    f.b1 = loadFragB(izBase, offB, k0);
    f.b2 = loadFragB(inBase, offB, k0);
    f.b3 = loadFragB(hrBase, offB, k0);
    f.b4 = loadFragB(hzBase, offB, k0);
    f.b5 = loadFragB(hnBase, offB, k0);
    return f;
}

// ---------------- prep kernels ----------------
__global__ void prep_weights_kernel(const float* Wih, const float* Whh, const float* Wout,
                                    __bf16* WihB, __bf16* WhhB, __bf16* WoutB) {
    int idx = blockIdx.x * blockDim.x + threadIdx.x;
    const int NW = N3 * HID;
    if (idx < NW) {
        WihB[idx] = (__bf16)Wih[idx];
        WhhB[idx] = (__bf16)Whh[idx];
    }
    if (idx < S_DIM * O_DIM * HID) {
        WoutB[idx] = (__bf16)Wout[idx];
    }
}

__global__ void init_state_kernel(float* hF, __bf16* hB, float* x, float* dxs) {
    int idx = blockIdx.x * blockDim.x + threadIdx.x;
    if (idx < MROWS * HID) { hF[idx] = 0.0f; hB[idx] = (__bf16)0.0f; }
    if (idx < BATCH * S_DIM) { x[idx] = 0.0f; dxs[idx] = 0.0f; }
}

// ---------------- per-step kernels ----------------
// 1) dynamics + innovation + normalized NN input  (1 block, 128 threads = batch)
__global__ void step_pre_kernel(const float* __restrict__ y_seq,
                                const float* __restrict__ x_state,
                                const float* __restrict__ dx_state,
                                float* __restrict__ x_pred,
                                float* __restrict__ innov,
                                float* __restrict__ nn_in, int t) {
    int b = threadIdx.x;
    float xp[S_DIM];
    for (int s = 0; s < S_DIM; ++s) {
        float xv = x_state[b * S_DIM + s];
        xp[s] = 0.9f * xv + 0.1f * sinf(xv);
        x_pred[b * S_DIM + s] = xp[s];
    }
    float iv[O_DIM]; float n2i = 0.0f;
    for (int o = 0; o < O_DIM; ++o) {
        float yh = tanhf(xp[o]);
        float d  = y_seq[(b * SEQ + t) * O_DIM + o] - yh;
        iv[o] = d;
        innov[b * O_DIM + o] = d;
        n2i += d * d;
    }
    float dxv[S_DIM]; float n2d = 0.0f;
    for (int s = 0; s < S_DIM; ++s) {
        dxv[s] = dx_state[b * S_DIM + s];
        n2d += dxv[s] * dxv[s];
    }
    float inv_d = 1.0f / fmaxf(sqrtf(n2d), 1e-12f);
    float inv_i = 1.0f / fmaxf(sqrtf(n2i), 1e-12f);
    for (int s = 0; s < S_DIM; ++s) nn_in[b * 12 + s] = dxv[s] * inv_d;
    for (int o = 0; o < O_DIM; ++o) nn_in[b * 12 + 8 + o] = iv[o] * inv_i;
}

// 2) input layer + relu + MC-dropout expansion to bf16 a_ens [NS*B, HID]
__global__ void act_kernel(const float* __restrict__ nn_in,
                           const float* __restrict__ W1,
                           const float* __restrict__ b1,
                           __bf16* __restrict__ aB, int t) {
    int idx = blockIdx.x * blockDim.x + threadIdx.x;
    if (idx >= BATCH * HID) return;
    int b = idx / HID;
    int h = idx - b * HID;
    float acc = b1[h];
    #pragma unroll
    for (int k = 0; k < 12; ++k) acc += nn_in[b * 12 + k] * W1[h * 12 + k];
    float a = fmaxf(acc, 0.0f);
    for (int j = 0; j < NS; ++j) {
        float mk = dropmask(0x1234u + (unsigned)t, (unsigned)j, (unsigned)idx);
        aB[(size_t)(j * BATCH + b) * HID + h] = (__bf16)(a * mk);
    }
}

#define GRU_WMMA6(F)                                                                                        \
    cir  = __builtin_amdgcn_wmma_f32_16x16x32_bf16(false, (F).fa, false, (F).b0, (short)0, cir,  false, false); \
    ciz  = __builtin_amdgcn_wmma_f32_16x16x32_bf16(false, (F).fa, false, (F).b1, (short)0, ciz,  false, false); \
    cin  = __builtin_amdgcn_wmma_f32_16x16x32_bf16(false, (F).fa, false, (F).b2, (short)0, cin,  false, false); \
    chr_ = __builtin_amdgcn_wmma_f32_16x16x32_bf16(false, (F).fh, false, (F).b3, (short)0, chr_, false, false); \
    chz  = __builtin_amdgcn_wmma_f32_16x16x32_bf16(false, (F).fh, false, (F).b4, (short)0, chz,  false, false); \
    chn  = __builtin_amdgcn_wmma_f32_16x16x32_bf16(false, (F).fh, false, (F).b5, (short)0, chn,  false, false)

// 3) fused GRU step: six WMMA accumulation chains per 16x16 output tile, gate math in epilogue.
//    Software-pipelined with 2x-unrolled ping-pong (no rotation copies):
//    body = { load f1@k ; compute f0 ; load f0@k+32 ; compute f1 }.
//    waves: mTile (128) x nTile (50) = 6400 ; blockDim=128 -> 1600 blocks
__global__ __launch_bounds__(128, 1)
void gru_wmma_kernel(const __bf16* __restrict__ aB,
                     const __bf16* __restrict__ hBin,
                     const float*  __restrict__ hFin,
                     const __bf16* __restrict__ WihB,
                     const __bf16* __restrict__ WhhB,
                     const float*  __restrict__ bih,
                     const float*  __restrict__ bhh,
                     float*  __restrict__ hFout,
                     __bf16* __restrict__ hBout) {
    int wave  = (blockIdx.x * blockDim.x + threadIdx.x) >> 5;
    int lane  = threadIdx.x & 31;
    int mTile = wave & 127;        // 0..127
    int nTile = wave >> 7;         // 0..49
    int m0 = mTile * 16;
    int j0 = nTile * 16;

    int r  = lane & 15;
    int hi = lane >> 4;
    int offA = r * HID + hi * 8;   // per-lane, loop-invariant (elements)
    int offB = r * HID + hi * 8;

    // wave-uniform tile bases (SGPR bases; gate-slab deltas + k0 fold into load immediates)
    const __bf16* aBase  = aB   + (size_t)m0 * HID;
    const __bf16* hBase  = hBin + (size_t)m0 * HID;
    const __bf16* irBase = WihB + (size_t)j0 * HID;
    const __bf16* izBase = irBase + (size_t)HID * HID;
    const __bf16* inBase = irBase + (size_t)(2 * HID) * HID;
    const __bf16* hrBase = WhhB + (size_t)j0 * HID;
    const __bf16* hzBase = hrBase + (size_t)HID * HID;
    const __bf16* hnBase = hrBase + (size_t)(2 * HID) * HID;

    v8f zero = {0.f, 0.f, 0.f, 0.f, 0.f, 0.f, 0.f, 0.f};
    v8f cir = zero, ciz = zero, cin = zero;
    v8f chr_ = zero, chz = zero, chn = zero;

    // ---- pipeline prologue: k = 0 ----
    FragSet f0 = load_all(aBase, hBase, irBase, izBase, inBase,
                          hrBase, hzBase, hnBase, offA, offB, 0);

    // 25 k-steps total: 1 prologue + 12 unrolled pairs + 1 tail compute
    #pragma unroll 1
    for (int k0 = 32; k0 < HID - 32; k0 += 64) {
        FragSet f1 = load_all(aBase, hBase, irBase, izBase, inBase,
                              hrBase, hzBase, hnBase, offA, offB, k0);
        GRU_WMMA6(f0);
        f0 = load_all(aBase, hBase, irBase, izBase, inBase,
                      hrBase, hzBase, hnBase, offA, offB, k0 + 32);
        GRU_WMMA6(f1);
    }

    // ---- pipeline tail: consume last fragment set (k = HID-32) ----
    GRU_WMMA6(f0);

    Acc Air, Aiz, Ain, Ahr, Ahz, Ahn;
    Air.v = cir; Aiz.v = ciz; Ain.v = cin;
    Ahr.v = chr_; Ahz.v = chz; Ahn.v = chn;

    int col = lane & 15;
    int rb  = (lane >> 4) * 8;
    int j   = j0 + col;
    float b_ir = bih[j],           b_hr = bhh[j];
    float b_iz = bih[HID + j],     b_hz = bhh[HID + j];
    float b_in = bih[2 * HID + j], b_hn = bhh[2 * HID + j];

    #pragma unroll
    for (int v = 0; v < 8; ++v) {
        int m = m0 + rb + v;
        float gir = Air.f[v] + b_ir;
        float giz = Aiz.f[v] + b_iz;
        float gin = Ain.f[v] + b_in;
        float ghr = Ahr.f[v] + b_hr;
        float ghz = Ahz.f[v] + b_hz;
        float ghn = Ahn.f[v] + b_hn;
        float rg = 1.0f / (1.0f + expf(-(gir + ghr)));
        float zg = 1.0f / (1.0f + expf(-(giz + ghz)));
        float ng = tanhf(gin + rg * ghn);
        float hp = hFin[(size_t)m * HID + j];
        float hn_ = (1.0f - zg) * ng + zg * hp;
        hFout[(size_t)m * HID + j] = hn_;
        hBout[(size_t)m * HID + j] = (__bf16)hn_;
    }
}

// 4) K_vec = h_new @ W_out^T + b_out, with dropout.  waves: 128 mTiles x 2 nTiles = 256 -> 64 blocks of 128
__global__ __launch_bounds__(128, 1)
void kvec_wmma_kernel(const __bf16* __restrict__ hB,
                      const __bf16* __restrict__ WoutB,
                      const float*  __restrict__ b_out,
                      float* __restrict__ Kmat, int t) {
    int wave = (blockIdx.x * blockDim.x + threadIdx.x) >> 5;
    int lane = threadIdx.x & 31;
    int mTile = wave >> 1;
    int n0 = (wave & 1) * 16;
    int m0 = mTile * 16;

    int r  = lane & 15;
    int hi = lane >> 4;
    int offA = r * HID + hi * 8;
    int offB = r * HID + hi * 8;
    const __bf16* aBase = hB + (size_t)m0 * HID;
    const __bf16* bBase = WoutB + (size_t)n0 * HID;

    v8f zero = {0.f, 0.f, 0.f, 0.f, 0.f, 0.f, 0.f, 0.f};
    v8f c = zero;

    // 2x-unrolled ping-pong pipeline
    v16bf fa0 = loadFragA(aBase, offA, 0);
    v16bf fb0 = loadFragB(bBase, offB, 0);
    #pragma unroll 1
    for (int k0 = 32; k0 < HID - 32; k0 += 64) {
        v16bf fa1 = loadFragA(aBase, offA, k0);
        v16bf fb1 = loadFragB(bBase, offB, k0);
        c = __builtin_amdgcn_wmma_f32_16x16x32_bf16(false, fa0, false, fb0, (short)0, c, false, false);
        fa0 = loadFragA(aBase, offA, k0 + 32);
        fb0 = loadFragB(bBase, offB, k0 + 32);
        c = __builtin_amdgcn_wmma_f32_16x16x32_bf16(false, fa1, false, fb1, (short)0, c, false, false);
    }
    c = __builtin_amdgcn_wmma_f32_16x16x32_bf16(false, fa0, false, fb0, (short)0, c, false, false);

    Acc A; A.v = c;
    int col = lane & 15;
    int rb  = (lane >> 4) * 8;
    int n   = n0 + col;
    float bo = b_out[n];
    #pragma unroll
    for (int v = 0; v < 8; ++v) {
        int m = m0 + rb + v;
        float val = A.f[v] + bo;
        float mk = dropmask(0x9876u + (unsigned)t, 1u, (unsigned)(m * 32 + n));
        Kmat[(size_t)m * 32 + n] = val * mk;
    }
}

// 5) ensemble update: mean, covariance, state/dx update, outputs.  1 block per batch elem, 128 threads
__global__ void finalize_kernel(const float* __restrict__ Kmat,
                                const float* __restrict__ x_pred,
                                const float* __restrict__ innov,
                                float* __restrict__ x_state,
                                float* __restrict__ dx_state,
                                float* __restrict__ out_xs,
                                float* __restrict__ out_Ps, int t) {
    int b = blockIdx.x;
    int tid = threadIdx.x;
    int j = tid >> 3;      // sample 0..15
    int s = tid & 7;       // state 0..7
    __shared__ float e[NS][S_DIM];
    __shared__ float xf[S_DIM];

    float acc = x_pred[b * S_DIM + s];
    const float* Krow = Kmat + ((size_t)(j * BATCH + b) * 32 + s * O_DIM);
    #pragma unroll
    for (int o = 0; o < O_DIM; ++o) acc += Krow[o] * innov[b * O_DIM + o];
    e[j][s] = acc;
    __syncthreads();

    if (tid < S_DIM) {
        float m = 0.0f;
        for (int jj = 0; jj < NS; ++jj) m += e[jj][tid];
        xf[tid] = m * (1.0f / NS);
    }
    __syncthreads();

    if (tid < S_DIM) {
        float xv = xf[tid];
        out_xs[((size_t)b * SEQ + t) * S_DIM + tid] = xv;
        x_state[b * S_DIM + tid] = xv;
        dx_state[b * S_DIM + tid] = xv - x_pred[b * S_DIM + tid];
    }
    if (tid < S_DIM * S_DIM) {
        int ss = tid >> 3, u = tid & 7;
        float a2 = 0.0f;
        for (int jj = 0; jj < NS; ++jj)
            a2 += (e[jj][ss] - xf[ss]) * (e[jj][u] - xf[u]);
        out_Ps[((size_t)b * SEQ + t) * 64 + tid] = a2 * (1.0f / NS);
    }
}

// ---------------- host orchestration ----------------
extern "C" void kernel_launch(void* const* d_in, const int* in_sizes, int n_in,
                              void* d_out, int out_size, void* d_ws, size_t ws_size,
                              hipStream_t stream) {
    const float* y_seq = (const float*)d_in[0];
    const float* W1    = (const float*)d_in[1];
    const float* b1    = (const float*)d_in[2];
    const float* W_ih  = (const float*)d_in[3];
    const float* W_hh  = (const float*)d_in[4];
    const float* b_ih  = (const float*)d_in[5];
    const float* b_hh  = (const float*)d_in[6];
    const float* W_out = (const float*)d_in[7];
    const float* b_out = (const float*)d_in[8];

    float* out_xs = (float*)d_out;                           // [B,T,S]
    float* out_Ps = out_xs + (size_t)BATCH * SEQ * S_DIM;    // [B,T,S,S]

    char* ws = (char*)d_ws;
    size_t off = 0;
    auto alloc = [&](size_t bytes) -> char* {
        char* p = ws + off;
        off = (off + bytes + 255) & ~(size_t)255;
        return p;
    };
    __bf16* WihB  = (__bf16*)alloc((size_t)N3 * HID * 2);
    __bf16* WhhB  = (__bf16*)alloc((size_t)N3 * HID * 2);
    __bf16* WoutB = (__bf16*)alloc((size_t)S_DIM * O_DIM * HID * 2);
    __bf16* aB    = (__bf16*)alloc((size_t)MROWS * HID * 2);
    float*  hF0   = (float*) alloc((size_t)MROWS * HID * 4);
    float*  hF1   = (float*) alloc((size_t)MROWS * HID * 4);
    __bf16* hB0   = (__bf16*)alloc((size_t)MROWS * HID * 2);
    __bf16* hB1   = (__bf16*)alloc((size_t)MROWS * HID * 2);
    float*  Kmat  = (float*) alloc((size_t)MROWS * 32 * 4);
    float*  x_st  = (float*) alloc((size_t)BATCH * S_DIM * 4);
    float*  dx_st = (float*) alloc((size_t)BATCH * S_DIM * 4);
    float*  x_pr  = (float*) alloc((size_t)BATCH * S_DIM * 4);
    float*  innov = (float*) alloc((size_t)BATCH * O_DIM * 4);
    float*  nn_in = (float*) alloc((size_t)BATCH * 12 * 4);

    {
        int nw = N3 * HID;
        prep_weights_kernel<<<(nw + 255) / 256, 256, 0, stream>>>(
            W_ih, W_hh, W_out, WihB, WhhB, WoutB);
        int nz = MROWS * HID;
        init_state_kernel<<<(nz + 255) / 256, 256, 0, stream>>>(hF0, hB0, x_st, dx_st);
    }

    for (int t = 0; t < SEQ; ++t) {
        float*  hFin  = (t & 1) ? hF1 : hF0;
        float*  hFout = (t & 1) ? hF0 : hF1;
        __bf16* hBin  = (t & 1) ? hB1 : hB0;
        __bf16* hBout = (t & 1) ? hB0 : hB1;

        step_pre_kernel<<<1, BATCH, 0, stream>>>(y_seq, x_st, dx_st, x_pr, innov, nn_in, t);

        act_kernel<<<(BATCH * HID + 255) / 256, 256, 0, stream>>>(nn_in, W1, b1, aB, t);

        gru_wmma_kernel<<<1600, 128, 0, stream>>>(aB, hBin, hFin, WihB, WhhB,
                                                  b_ih, b_hh, hFout, hBout);

        kvec_wmma_kernel<<<64, 128, 0, stream>>>(hBout, WoutB, b_out, Kmat, t);

        finalize_kernel<<<BATCH, 128, 0, stream>>>(Kmat, x_pr, innov, x_st, dx_st,
                                                   out_xs, out_Ps, t);
    }
}
